// Mixtape_82824149336212
// MI455X (gfx1250) — matrix operand order repaired and checked by reading, not compiled, not run on hardware
//
#include <hip/hip_runtime.h>
#include <hip/hip_bf16.h>
#include <math.h>

// ---------------- problem constants (from reference) ----------------
#define BDIM   2
#define TDIM   512
#define BT     1024          // B*T
#define D1     256
#define D2     128
#define DM     1024
#define VOC    32000
#define KG     4             // mixture gates
#define NSUB   2             // N sub-tiles per wave (16 cols each)

typedef __attribute__((ext_vector_type(16))) _Float16 v16h;
typedef __attribute__((ext_vector_type(8)))  _Float16 v8h;
typedef __attribute__((ext_vector_type(8)))  float    v8f;

static __device__ __forceinline__ v8f vzero8() {
    v8f z;
#pragma unroll
    for (int i = 0; i < 8; ++i) z[i] = 0.0f;
    return z;
}

static __device__ __forceinline__ v16h cat8(v8h lo, v8h hi) {
    v16h r;
#pragma unroll
    for (int i = 0; i < 8; ++i) { r[i] = lo[i]; r[8 + i] = hi[i]; }
    return r;
}

static __device__ __forceinline__ v8f wmma_f16(v16h a, v16h b, v8f c) {
    // D = A(16x32 f16) x B(32x16 f16) + C(16x16 f32)
    return __builtin_amdgcn_wmma_f32_16x16x32_f16(
        /*neg_a=*/false, a, /*neg_b=*/false, b,
        /*c_mod=*/(short)0, c, /*reuse_a=*/false, /*reuse_b=*/false);
}

static __device__ __forceinline__ float sigmoidf_fast(float x) {
    return 1.0f / (1.0f + __expf(-x));
}

// ------------------------------------------------------------------
// Kernel 1: per-token precompute
//   hc[bt][k][d]  = tanh( sum_i H[k][d][i] * gc[bt][i] )   (f16)
//   tU[bt][k][j]  = tanh( sum_i U[k][j][i] * gc[bt][i] )   (f16)
//   ug[bt][k]     =       sum_i u[k][i]    * gc[bt][i]     (f32)
// One block per bt row; gc row cached in LDS. ~2.4 GF total.
// ------------------------------------------------------------------
__global__ __launch_bounds__(256) void precompute_kernel(
    const float* __restrict__ gc, const float* __restrict__ H,
    const float* __restrict__ U,  const float* __restrict__ u,
    _Float16* __restrict__ hc, _Float16* __restrict__ tU,
    float* __restrict__ ug)
{
    __shared__ float g[D1];
    const int bt = blockIdx.x;
    g[threadIdx.x] = gc[(size_t)bt * D1 + threadIdx.x];
    __syncthreads();

    const float4* g4 = (const float4*)g;

    // hc: KG*DM = 4096 outputs
    for (int o = threadIdx.x; o < KG * DM; o += 256) {
        const int k = o / DM, d = o % DM;
        const float4* h4 = (const float4*)(H + ((size_t)k * DM + d) * D1);
        float s = 0.0f;
#pragma unroll 8
        for (int i = 0; i < D1 / 4; ++i) {
            float4 a = h4[i], b = g4[i];
            s += a.x * b.x + a.y * b.y + a.z * b.z + a.w * b.w;
        }
        hc[((size_t)bt * KG + k) * DM + d] = (_Float16)tanhf(s);
    }

    // tU: KG*D2 = 512 outputs
    for (int o = threadIdx.x; o < KG * D2; o += 256) {
        const int k = o / D2, j = o % D2;
        const float4* u4 = (const float4*)(U + ((size_t)k * D2 + j) * D1);
        float s = 0.0f;
#pragma unroll 8
        for (int i = 0; i < D1 / 4; ++i) {
            float4 a = u4[i], b = g4[i];
            s += a.x * b.x + a.y * b.y + a.z * b.z + a.w * b.w;
        }
        tU[((size_t)bt * KG + k) * D2 + j] = (_Float16)tanhf(s);
    }

    // ug: KG outputs
    if (threadIdx.x < KG) {
        const int k = threadIdx.x;
        const float4* u4 = (const float4*)(u + (size_t)k * D1);
        float s = 0.0f;
#pragma unroll 8
        for (int i = 0; i < D1 / 4; ++i) {
            float4 a = u4[i], b = g4[i];
            s += a.x * b.x + a.y * b.y + a.z * b.z + a.w * b.w;
        }
        ug[(size_t)bt * KG + k] = s;
    }
}

// ------------------------------------------------------------------
// Kernel 2: f32 -> f16 transpose:  src[R][C] -> dst[C][R]
// (emb -> embT[DM][V], v -> vT[D2][V]; makes WMMA B-fragment loads
//  contiguous across lanes: lane = N col, 16 K rows per lane.)
// ------------------------------------------------------------------
__global__ __launch_bounds__(256) void transpose_f32_to_f16(
    const float* __restrict__ src, _Float16* __restrict__ dst,
    int R, int C)
{
    __shared__ _Float16 tile[32][33];
    const int bx = blockIdx.x * 32;   // column base (C)
    const int by = blockIdx.y * 32;   // row base (R)
    const int tx = threadIdx.x & 31;
    const int ty = threadIdx.x >> 5;  // 0..7
#pragma unroll
    for (int i = 0; i < 32; i += 8)
        tile[ty + i][tx] = (_Float16)src[(size_t)(by + ty + i) * C + bx + tx];
    __syncthreads();
#pragma unroll
    for (int i = 0; i < 32; i += 8)
        dst[(size_t)(bx + ty + i) * R + by + tx] = tile[tx][ty + i];
}

// ------------------------------------------------------------------
// Kernel 3: fused mixture head.
// Block tile: 16 (BT rows) x 256 (V cols); 8 waves, each owning a
// 16x32 tile (NSUB=2 x 16x16 WMMA sub-tiles). Per wave per 32-K
// chunk each gate's A fragment (shared across sub-tiles) is loaded
// once and feeds 2 WMMAs (8 per chunk) -> ~2x better operand-bandwidth amortization
// than a 16x16 wave tile.
//   lc_k (k<3) = tU_k(16xD2) x vT(D2x32)          24 WMMAs
//   pi         = sigmoid-tree(lc + ug + b)        (elementwise)
//   dots_k     = hc_k(16xDM) x embT(DMx32)        256 WMMAs
//   logits     = sum_k pi_k * dots_k  -> d_out
// ------------------------------------------------------------------
__global__ __launch_bounds__(256) void mixtape_logits_kernel(
    const _Float16* __restrict__ hcb,   // [BT][KG][DM]
    const _Float16* __restrict__ tub,   // [BT][KG][D2]
    const float*    __restrict__ ug,    // [BT][KG]
    const float*    __restrict__ bvk,   // [VOC][KG]
    const _Float16* __restrict__ embT,  // [DM][VOC]
    const _Float16* __restrict__ vT,    // [D2][VOC]
    float* __restrict__ logits)         // [BT][VOC]
{
    __shared__ float ugs[16 * KG];      // ug rows for this M strip

    const int wave = threadIdx.x >> 5;
    const int lane = threadIdx.x & 31;
    const int m0   = blockIdx.y * 16;
    const int n0   = blockIdx.x * (NSUB * 16 * 8) + wave * (NSUB * 16);
    const int mrow = lane & 15;   // A-fragment row for this lane
    const int sel  = lane >> 4;   // lane half-group (K-chunk select)
    const int ncol = lane & 15;   // C/D column for this lane

    if (threadIdx.x < 16 * KG)
        ugs[threadIdx.x] = ug[(size_t)m0 * KG + threadIdx.x];
    __syncthreads();

    // ---- lc accumulation (gates 0..2 only; gate 3 unused by gamma) ----
    v8f lcacc[3][NSUB];
#pragma unroll
    for (int k = 0; k < 3; ++k)
#pragma unroll
        for (int t = 0; t < NSUB; ++t) lcacc[k][t] = vzero8();

#pragma unroll
    for (int jc = 0; jc < D2; jc += 32) {
        v16h bf[NSUB];
#pragma unroll
        for (int t = 0; t < NSUB; ++t)
            for (int s = 0; s < 16; ++s)
                bf[t][s] = vT[(size_t)(jc + sel * 8 + (s & 7) + (s >> 3) * 16) * VOC + n0 + 16 * t + ncol];
        const _Float16* ab =
            tub + ((size_t)(m0 + mrow) * KG) * D2 + jc + sel * 8;
#pragma unroll
        for (int k = 0; k < 3; ++k) {
            const v16h a =
                cat8(*(const v8h*)(ab + k * D2), *(const v8h*)(ab + k * D2 + 16));
#pragma unroll
            for (int t = 0; t < NSUB; ++t)
                lcacc[k][t] = wmma_f16(a, bf[t], lcacc[k][t]);
        }
    }

    // ---- sigmoid tree -> mixture priors ----
    v8f piv[KG][NSUB];
#pragma unroll
    for (int t = 0; t < NSUB; ++t) {
        // b[v, 0..3] as a single b128 load
        const float4 bq = *(const float4*)(bvk + (size_t)(n0 + 16 * t + ncol) * KG);
#pragma unroll
        for (int r = 0; r < 8; ++r) {
            const int mi = r + sel * 8;          // row within M strip
            const float l0 = lcacc[0][t][r] + ugs[mi * KG + 0] + bq.x;
            const float l1 = lcacc[1][t][r] + ugs[mi * KG + 1] + bq.y;
            const float l2 = lcacc[2][t][r] + ugs[mi * KG + 2] + bq.z;
            const float g0 = sigmoidf_fast(l0);
            const float g1 = sigmoidf_fast(l1);
            const float g2 = sigmoidf_fast(l2);
            piv[0][t][r] = g0 * g1;
            piv[1][t][r] = g0 * (1.0f - g1);
            piv[2][t][r] = (1.0f - g0) * g2;
            piv[3][t][r] = (1.0f - g0) * (1.0f - g2);
        }
    }

    // ---- dots accumulation: hc x embT (dominant 268 GF GEMM) ----
    v8f dacc[KG][NSUB];
#pragma unroll
    for (int k = 0; k < KG; ++k)
#pragma unroll
        for (int t = 0; t < NSUB; ++t) dacc[k][t] = vzero8();

    for (int dc = 0; dc < DM; dc += 32) {
        // speculative prefetch of the emb stream 2 chunks ahead
        if (dc + 64 < DM)
            __builtin_prefetch(embT + (size_t)(dc + 64 + lane) * VOC + n0, 0, 0);

        v16h bf[NSUB];
#pragma unroll
        for (int t = 0; t < NSUB; ++t)
            for (int s = 0; s < 16; ++s)
                bf[t][s] = embT[(size_t)(dc + sel * 8 + (s & 7) + (s >> 3) * 16) * VOC + n0 + 16 * t + ncol];
        const _Float16* ab =
            hcb + ((size_t)(m0 + mrow) * KG) * DM + dc + sel * 8;
#pragma unroll
        for (int k = 0; k < KG; ++k) {
            const v16h a =
                cat8(*(const v8h*)(ab + k * DM), *(const v8h*)(ab + k * DM + 16));
#pragma unroll
            for (int t = 0; t < NSUB; ++t)
                dacc[k][t] = wmma_f16(a, bf[t], dacc[k][t]);
        }
    }

    // ---- combine and store logits ----
#pragma unroll
    for (int t = 0; t < NSUB; ++t) {
#pragma unroll
        for (int r = 0; r < 8; ++r) {
            const int m = m0 + r + sel * 8;
            logits[(size_t)m * VOC + n0 + 16 * t + ncol] =
                piv[0][t][r] * dacc[0][t][r] + piv[1][t][r] * dacc[1][t][r] +
                piv[2][t][r] * dacc[2][t][r] + piv[3][t][r] * dacc[3][t][r];
        }
    }
}

// ------------------------------------------------------------------
// Kernel 4: in-place row softmax over V (block per BT row).
// ------------------------------------------------------------------
__global__ __launch_bounds__(256) void softmax_rows_kernel(float* __restrict__ x)
{
    const int row  = blockIdx.x;
    float* p = x + (size_t)row * VOC;
    const int lane = threadIdx.x & 31;
    const int wave = threadIdx.x >> 5;
    __shared__ float red[8];
    __shared__ float bcast;

    float m = -3.402823466e38f;
    for (int i = threadIdx.x; i < VOC; i += 256) m = fmaxf(m, p[i]);
#pragma unroll
    for (int off = 16; off > 0; off >>= 1) m = fmaxf(m, __shfl_xor(m, off));
    if (lane == 0) red[wave] = m;
    __syncthreads();
    if (threadIdx.x == 0) {
        float mm = red[0];
#pragma unroll
        for (int i = 1; i < 8; ++i) mm = fmaxf(mm, red[i]);
        bcast = mm;
    }
    __syncthreads();
    const float rowmax = bcast;

    float s = 0.0f;
    for (int i = threadIdx.x; i < VOC; i += 256) {
        const float e = __expf(p[i] - rowmax);
        p[i] = e;
        s += e;
    }
#pragma unroll
    for (int off = 16; off > 0; off >>= 1) s += __shfl_xor(s, off);
    if (lane == 0) red[wave] = s;
    __syncthreads();
    if (threadIdx.x == 0) {
        float ss = 0.0f;
#pragma unroll
        for (int i = 0; i < 8; ++i) ss += red[i];
        bcast = ss;
    }
    __syncthreads();
    const float inv = 1.0f / bcast;
    for (int i = threadIdx.x; i < VOC; i += 256) p[i] *= inv;
}

// ------------------------------------------------------------------
// launcher
// ------------------------------------------------------------------
extern "C" void kernel_launch(void* const* d_in, const int* in_sizes, int n_in,
                              void* d_out, int out_size, void* d_ws, size_t ws_size,
                              hipStream_t stream)
{
    (void)in_sizes; (void)n_in; (void)out_size; (void)ws_size;

    const float* gc  = (const float*)d_in[0];  // (B,T,D1)
    const float* H   = (const float*)d_in[1];  // (K,DM,D1)
    const float* U   = (const float*)d_in[2];  // (K,D2,D1)
    const float* v   = (const float*)d_in[3];  // (V,D2)
    const float* u   = (const float*)d_in[4];  // (K,D1)
    const float* b   = (const float*)d_in[5];  // (V,K)
    const float* emb = (const float*)d_in[6];  // (V,DM)
    float* out = (float*)d_out;                // (B,T,V)

    // workspace carve-out (~81 MB)
    size_t off = 0;
    char* ws = (char*)d_ws;
    auto take = [&](size_t bytes) -> void* {
        void* p = ws + off;
        off += (bytes + 255) & ~(size_t)255;
        return p;
    };
    _Float16* embT = (_Float16*)take((size_t)DM * VOC * sizeof(_Float16));
    _Float16* vT   = (_Float16*)take((size_t)D2 * VOC * sizeof(_Float16));
    _Float16* hcb  = (_Float16*)take((size_t)BT * KG * DM * sizeof(_Float16));
    _Float16* tub  = (_Float16*)take((size_t)BT * KG * D2 * sizeof(_Float16));
    float*    ugb  = (float*)take((size_t)BT * KG * sizeof(float));

    // 1) per-token precompute
    precompute_kernel<<<BT, 256, 0, stream>>>(gc, H, U, u, hcb, tub, ugb);

    // 2) transpose+convert emb and v to f16, K-major for WMMA B frags
    transpose_f32_to_f16<<<dim3(DM / 32, VOC / 32), 256, 0, stream>>>(emb, embT, VOC, DM);
    transpose_f32_to_f16<<<dim3(D2 / 32, VOC / 32), 256, 0, stream>>>(v, vT, VOC, D2);

    // 3) fused WMMA mixture head -> logits in d_out
    //    block tile: 16 (M) x 256 (N); grid = (V/256, BT/16)
    mixtape_logits_kernel<<<dim3(VOC / (NSUB * 16 * 8), BT / 16), 256, 0, stream>>>(
        hcb, tub, ugb, b, embT, vT, out);

    // 4) row softmax over V, in place
    softmax_rows_kernel<<<BT, 256, 0, stream>>>(out);
}
